// LayerNormGRUCell_4561255268489
// MI455X (gfx1250) — matrix operand order repaired and checked by reading, not compiled, not run on hardware
//
#include <hip/hip_runtime.h>
#include <hip/hip_bf16.h>

// ---------------------------------------------------------------------------
// LayerNorm-GRU cell, fused, for MI455X (gfx1250, wave32, WMMA).
//   B = 32768, I = H = 512.
//   GEMMs on v_wmma_f32_16x16x32_bf16 (f32 accumulate), M=32 rows/block,
//   A fragments register-resident (2 row-tiles share every B fragment),
//   bf16 LDS scoreboard (192 KB) + fused (vectorized) LN/sigmoid/tanh epilogue.
// ---------------------------------------------------------------------------

typedef __bf16 v16bf __attribute__((ext_vector_type(16)));
typedef __bf16 v8bf  __attribute__((ext_vector_type(8)));
typedef float  v8f   __attribute__((ext_vector_type(8)));

static constexpr int kI    = 512;            // K dimension (input & hidden dim)
static constexpr int kH    = 512;            // hidden dim
static constexpr int kN    = 1536;           // concat columns: [2H | H]
static constexpr int kRows = 32;             // M rows per block (2 row-tiles)
static constexpr int kLdsAcc = kRows * 2 * kN * 2;     // bf16 scoreboard bytes
static constexpr int kLdsA   = kRows * kI * 2;         // A staging bytes

__device__ __forceinline__ __bf16 f2bf(float f) {
  unsigned u = __builtin_bit_cast(unsigned, f);
  unsigned r = u + 0x7FFFu + ((u >> 16) & 1u);   // round-to-nearest-even
  unsigned short s = (unsigned short)(r >> 16);
  return __builtin_bit_cast(__bf16, s);
}

__device__ __forceinline__ unsigned pack2bf(float a, float b) {
  unsigned ua = __builtin_bit_cast(unsigned, a);
  unsigned ub = __builtin_bit_cast(unsigned, b);
  ua = (ua + 0x7FFFu + ((ua >> 16) & 1u)) >> 16;
  ub = (ub + 0x7FFFu + ((ub >> 16) & 1u)) >> 16;
  return (ub << 16) | (ua & 0xFFFFu);
}

// Load two adjacent bf16 scoreboard entries (4-byte aligned) as floats.
__device__ __forceinline__ float2 ld2bf(const __bf16* p) {
  unsigned u = *(const unsigned*)p;
  float2 r;
  r.x = __builtin_bit_cast(float, u << 16);
  r.y = __builtin_bit_cast(float, u & 0xFFFF0000u);
  return r;
}

__device__ __forceinline__ float wave_sum(float v) {
#pragma unroll
  for (int off = 16; off > 0; off >>= 1) v += __shfl_xor(v, off, 32);
  return v;
}

// Build bf16, transposed, concatenated weight matrices in workspace:
//   WtX[n][k] = (n < 1024 ? W_i2h[k][n] : W_hatW[k][n-1024])
//   WtH[n][k] = (n < 1024 ? W_h2h[k][n] : W_hatU[k][n-1024])
__global__ void __launch_bounds__(256) gru_prep_weights(
    const float* __restrict__ Wi2h, const float* __restrict__ Wh2h,
    const float* __restrict__ WhatW, const float* __restrict__ WhatU,
    __bf16* __restrict__ WtX, __bf16* __restrict__ WtH) {
  const int total = 2 * kN * kI;
  int idx = blockIdx.x * 256 + threadIdx.x;
  if (idx >= total) return;
  const int per = kN * kI;
  int mat = idx / per;
  int rem = idx - mat * per;
  int k = rem / kN;          // consecutive threads -> consecutive n (coalesced reads)
  int n = rem - k * kN;
  float v;
  if (mat == 0) v = (n < 1024) ? Wi2h[(size_t)k * 1024 + n] : WhatW[(size_t)k * 512 + (n - 1024)];
  else          v = (n < 1024) ? Wh2h[(size_t)k * 1024 + n] : WhatU[(size_t)k * 512 + (n - 1024)];
  (mat == 0 ? WtX : WtH)[(size_t)n * kI + k] = f2bf(v);
}

__global__ void __launch_bounds__(128) gru_fused(
    const float* __restrict__ x, const float* __restrict__ h,
    const __bf16* __restrict__ WtX, const __bf16* __restrict__ WtH,
    const float* __restrict__ b_i2h, const float* __restrict__ b_h2h,
    const float* __restrict__ b_hatW, const float* __restrict__ b_hatU,
    float* __restrict__ out) {
  extern __shared__ char smem_raw[];
  __bf16* sAcc = (__bf16*)smem_raw;               // [32][3072] bf16 pre-acts
  __bf16* sA   = (__bf16*)(smem_raw + kLdsAcc);   // [32][512]  bf16 A stage

  const int tid     = threadIdx.x;
  const int lane    = tid & 31;
  const int wave    = tid >> 5;       // 4 waves
  const int rowBase = blockIdx.x * kRows;
  const int mr      = lane & 15;      // row (A) / col (B/C) within tile
  const int half    = lane >> 4;      // lane-group selector per ISA layouts

  // ------------------- GEMM phase: sel 0 = x-side, sel 1 = h-side ----------
#pragma unroll 1
  for (int sel = 0; sel < 2; ++sel) {
    const float*  src = sel ? h : x;
    const __bf16* Wt  = sel ? WtH : WtX;

    // Stage 32 rows of A into LDS as bf16: float4 in, packed uint2 out.
    {
      const float4* s4 = (const float4*)(src + (size_t)rowBase * kI);
      uint2* d2 = (uint2*)sA;
      for (int i = tid; i < kRows * kI / 4; i += 128) {
        float4 v = s4[i];
        uint2 p;
        p.x = pack2bf(v.x, v.y);
        p.y = pack2bf(v.z, v.w);
        d2[i] = p;
      }
    }
    __syncthreads();

    // A fragments for both row-tiles, register resident for the whole N loop.
    // A 16x32 bf16 layout: lane(mr,half) holds K = {half*8..+7} U {16+half*8..+7}
    v16bf aF0[16], aF1[16];
    {
      const __bf16* a0 = sA + (size_t)mr * kI;
      const __bf16* a1 = sA + (size_t)(16 + mr) * kI;
#pragma unroll
      for (int kc = 0; kc < kI / 32; ++kc) {
        const int ka0 = kc * 32 + half * 8;
        union { v16bf v; v8bf hh[2]; } A;
        A.hh[0] = *(const v8bf*)(a0 + ka0);
        A.hh[1] = *(const v8bf*)(a0 + ka0 + 16);
        aF0[kc] = A.v;
        A.hh[0] = *(const v8bf*)(a1 + ka0);
        A.hh[1] = *(const v8bf*)(a1 + ka0 + 16);
        aF1[kc] = A.v;
      }
    }

#pragma unroll 1
    for (int t = wave; t < kN / 16; t += 4) {
      const int n0 = t * 16;
      const __bf16* bCol = Wt + (size_t)(n0 + mr) * kI;
      if (t + 4 < kN / 16)  // speculative L2 prefetch of next tile's B column
        __builtin_prefetch(Wt + (size_t)(n0 + 64 + mr) * kI, 0, 1);

      v8f acc0 = {}, acc1 = {};
#pragma unroll
      for (int kc = 0; kc < kI / 32; ++kc) {
        // B 32x16 bf16: lane(mr,half) holds K = {half*8..+7} U {16+half*8..+7}
        const int kb0 = kc * 32 + half * 8;
        union { v16bf v; v8bf hh[2]; } B;
        B.hh[0] = *(const v8bf*)(bCol + kb0);
        B.hh[1] = *(const v8bf*)(bCol + kb0 + 16);
        acc0 = __builtin_amdgcn_wmma_f32_16x16x32_bf16(
            false, aF0[kc], false, B.v, (short)0, acc0, false, false);
        acc1 = __builtin_amdgcn_wmma_f32_16x16x32_bf16(
            false, aF1[kc], false, B.v, (short)0, acc1, false, false);
      }
      // C/D 16x16 f32: lane col = n0+mr, VGPR v -> row half*8+v
      __bf16* dst = sAcc + (size_t)sel * kN + n0 + mr;
      const int rb = half * 8;
#pragma unroll
      for (int v = 0; v < 8; ++v) {
        dst[(size_t)(rb + v) * (2 * kN)]        = f2bf(acc0[v]);
        dst[(size_t)(16 + rb + v) * (2 * kN)]   = f2bf(acc1[v]);
      }
    }
    __syncthreads();
  }

  // ------------------- Fused LN + gates epilogue (8 rows per wave) ---------
  // Vectorized: 2 columns per lane-iteration (packed bf16 reads, float2 IO).
#pragma unroll 1
  for (int rr = 0; rr < 8; ++rr) {
    const int r    = wave * 8 + rr;
    const int grow = rowBase + r;
    const __bf16* ax = sAcc + (size_t)r * (2 * kN); // x@[W_i2h|W_hatW] row
    const __bf16* ah = ax + kN;                     // h@[W_h2h|W_hatU] row

    float s1 = 0.f, q1 = 0.f, s2 = 0.f, q2 = 0.f;
    for (int j = lane * 2; j < 1024; j += 64) {
      float2 a  = ld2bf(ax + j);
      float2 ba = *(const float2*)(b_i2h + j);
      float a0 = a.x + ba.x, a1 = a.y + ba.y;
      s1 += a0 + a1; q1 += a0 * a0 + a1 * a1;
      float2 b  = ld2bf(ah + j);
      float2 bb = *(const float2*)(b_h2h + j);
      float b0 = b.x + bb.x, b1 = b.y + bb.y;
      s2 += b0 + b1; q2 += b0 * b0 + b1 * b1;
    }
    float s3 = 0.f, q3 = 0.f, s4 = 0.f, q4 = 0.f;
    for (int j = lane * 2; j < 512; j += 64) {
      float2 a  = ld2bf(ax + 1024 + j);
      float2 ba = *(const float2*)(b_hatW + j);
      float a0 = a.x + ba.x, a1 = a.y + ba.y;
      s3 += a0 + a1; q3 += a0 * a0 + a1 * a1;
      float2 b  = ld2bf(ah + 1024 + j);
      float2 bb = *(const float2*)(b_hatU + j);
      float b0 = b.x + bb.x, b1 = b.y + bb.y;
      s4 += b0 + b1; q4 += b0 * b0 + b1 * b1;
    }
    s1 = wave_sum(s1); q1 = wave_sum(q1);
    s2 = wave_sum(s2); q2 = wave_sum(q2);
    s3 = wave_sum(s3); q3 = wave_sum(q3);
    s4 = wave_sum(s4); q4 = wave_sum(q4);

    const float mu1 = s1 * (1.f / 1024.f);
    const float rs1 = rsqrtf(q1 * (1.f / 1024.f) - mu1 * mu1 + 1e-5f);
    const float mu2 = s2 * (1.f / 1024.f);
    const float rs2 = rsqrtf(q2 * (1.f / 1024.f) - mu2 * mu2 + 1e-5f);
    const float mu3 = s3 * (1.f / 512.f);
    const float rs3 = rsqrtf(q3 * (1.f / 512.f) - mu3 * mu3 + 1e-5f);
    const float mu4 = s4 * (1.f / 512.f);
    const float rs4 = rsqrtf(q4 * (1.f / 512.f) - mu4 * mu4 + 1e-5f);

    for (int j = lane * 2; j < 512; j += 64) {
      float2 axz = ld2bf(ax + j),        bz1 = *(const float2*)(b_i2h + j);
      float2 ahz = ld2bf(ah + j),        bz2 = *(const float2*)(b_h2h + j);
      float2 axr = ld2bf(ax + 512 + j),  br1 = *(const float2*)(b_i2h + 512 + j);
      float2 ahr = ld2bf(ah + 512 + j),  br2 = *(const float2*)(b_h2h + 512 + j);
      float2 axw = ld2bf(ax + 1024 + j), bw  = *(const float2*)(b_hatW + j);
      float2 ahu = ld2bf(ah + 1024 + j), bu  = *(const float2*)(b_hatU + j);
      float2 hv  = *(const float2*)(h + (size_t)grow * kH + j);

      float2 res;
      {
        float g1 = (axz.x + bz1.x - mu1) * rs1 + (ahz.x + bz2.x - mu2) * rs2;
        float g2 = (axr.x + br1.x - mu1) * rs1 + (ahr.x + br2.x - mu2) * rs2;
        float z  = 1.f / (1.f + __expf(-g1));
        float rg = 1.f / (1.f + __expf(-g2));
        float hh = tanhf((axw.x + bw.x - mu3) * rs3 + rg * ((ahu.x + bu.x - mu4) * rs4));
        res.x = (1.f - z) * hv.x + z * hh;
      }
      {
        float g1 = (axz.y + bz1.y - mu1) * rs1 + (ahz.y + bz2.y - mu2) * rs2;
        float g2 = (axr.y + br1.y - mu1) * rs1 + (ahr.y + br2.y - mu2) * rs2;
        float z  = 1.f / (1.f + __expf(-g1));
        float rg = 1.f / (1.f + __expf(-g2));
        float hh = tanhf((axw.y + bw.y - mu3) * rs3 + rg * ((ahu.y + bu.y - mu4) * rs4));
        res.y = (1.f - z) * hv.y + z * hh;
      }
      *(float2*)(out + (size_t)grow * kH + j) = res;
    }
  }
}

extern "C" void kernel_launch(void* const* d_in, const int* in_sizes, int n_in,
                              void* d_out, int out_size, void* d_ws, size_t ws_size,
                              hipStream_t stream) {
  const float* x      = (const float*)d_in[0];
  const float* h      = (const float*)d_in[1];
  const float* W_i2h  = (const float*)d_in[2];
  const float* b_i2h  = (const float*)d_in[3];
  const float* W_h2h  = (const float*)d_in[4];
  const float* b_h2h  = (const float*)d_in[5];
  const float* W_hatW = (const float*)d_in[6];
  const float* b_hatW = (const float*)d_in[7];
  const float* W_hatU = (const float*)d_in[8];
  const float* b_hatU = (const float*)d_in[9];
  float* out = (float*)d_out;

  __bf16* WtX = (__bf16*)d_ws;                   // 1536x512 bf16 = 1.5 MB
  __bf16* WtH = WtX + (size_t)kN * kI;           // another 1.5 MB

  const int nB = in_sizes[0] / kI;               // 32768 rows

  {
    const int total = 2 * kN * kI;
    gru_prep_weights<<<(total + 255) / 256, 256, 0, stream>>>(
        W_i2h, W_h2h, W_hatW, W_hatU, WtX, WtH);
  }
  {
    const size_t lds = (size_t)kLdsAcc + (size_t)kLdsA;  // 224 KB
    gru_fused<<<nB / kRows, 128, lds, stream>>>(
        x, h, WtX, WtH, b_i2h, b_h2h, b_hatW, b_hatU, out);
  }
}
